// ROIToNetworkPool_45543833206851
// MI455X (gfx1250) — matrix operand order repaired and checked by reading, not compiled, not run on hardware
//
#include <hip/hip_runtime.h>
#include <hip/hip_bf16.h>

#define N_ROI   262144
#define DIM     256
#define HID     64
#define NNET    128
#define LDA     264            // padded bf16 row stride in LDS (256 + 8)
#define ROWS_PER 2048          // rows per scatter block

typedef __attribute__((ext_vector_type(16))) __bf16        v16bf;
typedef __attribute__((ext_vector_type(8)))  float         v8f;
typedef __attribute__((ext_vector_type(4)))  unsigned int  v4u;

union Frag16 {
    v16bf        b;
    v4u          q[2];
    unsigned int w[8];
};

static __device__ __forceinline__ unsigned hi_bits(unsigned u) { return u & 0xFFFF0000u; }

// ---------------------------------------------------------------------------
// Kernel 0: zero segment accumulators (deterministic per-call init)
// ---------------------------------------------------------------------------
__global__ __launch_bounds__(256) void init_kernel(unsigned* keys, float* denom, float* num)
{
    int i = blockIdx.x * 256 + threadIdx.x;
    if (i < NNET) { keys[i] = 0u; denom[i] = 0.0f; }
    if (i < NNET * DIM) num[i] = 0.0f;
}

// ---------------------------------------------------------------------------
// Kernel 1: fused  score = relu(x@W1 + b1) @ W2 + b2   via bf16x3 WMMA
//   block = 256 threads = 8 waves = 2 row-groups(16 rows) x 4 N-tiles(16 cols)
//   grid-strided over 8192 slabs of 32 rows; A fragments double-buffered in
//   registers so ds_load_b128 overlaps the previous chunk's WMMAs
// ---------------------------------------------------------------------------
__global__ __launch_bounds__(256) void mlp_score_kernel(
    const float* __restrict__ x,  const float* __restrict__ W1,
    const float* __restrict__ b1, const float* __restrict__ W2,
    const float* __restrict__ b2, float* __restrict__ score)
{
    __shared__ unsigned short sAhi[32 * LDA];
    __shared__ unsigned short sAlo[32 * LDA];
    __shared__ float          sScore[32];

    const int t    = threadIdx.x;
    const int lane = t & 31;
    const int wave = t >> 5;
    const int nt   = wave & 3;                 // N-tile of this wave
    const int rg   = wave >> 2;                // row-group (0/1)
    const int col  = nt * 16 + (lane & 15);    // hidden column this lane owns
    const int kofB = (lane < 16) ? 0 : 16;     // B fragment K offset (16-bit B 32x16)
    const int kofA = (lane < 16) ? 0 : 16;     // A fragment K offset (16-bit A 16x32)

    const float w2v = W2[col];
    const float b1v = b1[col];
    const float b2v = b2[0];

    // ---- W1 fragments, hi/lo bf16 split, resident in VGPRs (reused all slabs)
    Frag16 Bhi[8], Blo[8];
    #pragma unroll
    for (int kc = 0; kc < 8; ++kc) {
        #pragma unroll
        for (int v = 0; v < 8; ++v) {
            const int k0 = kc * 32 + kofB + 2 * v;
            const float f0 = W1[k0 * HID + col];
            const float f1 = W1[(k0 + 1) * HID + col];
            const unsigned h0 = hi_bits(__float_as_uint(f0));
            const unsigned h1 = hi_bits(__float_as_uint(f1));
            const float l0 = f0 - __uint_as_float(h0);
            const float l1 = f1 - __uint_as_float(h1);
            Bhi[kc].w[v] = (h0 >> 16) | h1;
            Blo[kc].w[v] = (__float_as_uint(l0) >> 16) | hi_bits(__float_as_uint(l1));
        }
    }

    const int rloc = rg * 16 + (lane & 15);    // LDS row this lane reads

    for (int slab = blockIdx.x; slab < N_ROI / 32; slab += gridDim.x) {
        __syncthreads();                        // previous slab fully consumed
        if (t < 32) sScore[t] = 0.0f;

        // ---- stage 32x256 fp32 tile -> packed bf16 hi/lo in LDS (coalesced)
        const float* xs = x + (size_t)slab * 32 * DIM;
        #pragma unroll
        for (int it = 0; it < 16; ++it) {
            const int e = it * 512 + t * 2;     // even element index
            const int r = e >> 8, c = e & 255;
            const float2 f = *(const float2*)(xs + e);
            const unsigned h0 = hi_bits(__float_as_uint(f.x));
            const unsigned h1 = hi_bits(__float_as_uint(f.y));
            const float l0 = f.x - __uint_as_float(h0);
            const float l1 = f.y - __uint_as_float(h1);
            *(unsigned*)&sAhi[r * LDA + c] = (h0 >> 16) | h1;
            *(unsigned*)&sAlo[r * LDA + c] =
                (__float_as_uint(l0) >> 16) | hi_bits(__float_as_uint(l1));
        }
        __syncthreads();

        // ---- 16x16 tile GEMM over K=256, 3 WMMAs per 32-K chunk, C init = b1
        v8f acc;
        #pragma unroll
        for (int v = 0; v < 8; ++v) acc[v] = b1v;

        Frag16 Ahi, Alo, AhiN, AloN;
        {
            const v4u* ph = (const v4u*)&sAhi[rloc * LDA + kofA];
            const v4u* pl = (const v4u*)&sAlo[rloc * LDA + kofA];
            Ahi.q[0] = ph[0]; Ahi.q[1] = ph[1];   // K = kofA..+15
            Alo.q[0] = pl[0]; Alo.q[1] = pl[1];
        }
        #pragma unroll
        for (int kc = 0; kc < 8; ++kc) {
            if (kc < 7) {                         // prefetch next chunk's fragments
                const v4u* ph = (const v4u*)&sAhi[rloc * LDA + (kc + 1) * 32 + kofA];
                const v4u* pl = (const v4u*)&sAlo[rloc * LDA + (kc + 1) * 32 + kofA];
                AhiN.q[0] = ph[0]; AhiN.q[1] = ph[1];
                AloN.q[0] = pl[0]; AloN.q[1] = pl[1];
            }
            acc = __builtin_amdgcn_wmma_f32_16x16x32_bf16(
                false, Ahi.b, false, Bhi[kc].b, (short)0, acc, false, false);
            acc = __builtin_amdgcn_wmma_f32_16x16x32_bf16(
                false, Ahi.b, false, Blo[kc].b, (short)0, acc, false, false);
            acc = __builtin_amdgcn_wmma_f32_16x16x32_bf16(
                false, Alo.b, false, Bhi[kc].b, (short)0, acc, false, false);
            if (kc < 7) { Ahi = AhiN; Alo = AloN; }
        }

        // ---- relu(h) * W2, reduce across 16-lane groups (rows stay separate)
        #pragma unroll
        for (int v = 0; v < 8; ++v) {
            float p = fmaxf(acc[v], 0.0f) * w2v;
            p += __shfl_xor(p, 1, 32);
            p += __shfl_xor(p, 2, 32);
            p += __shfl_xor(p, 4, 32);
            p += __shfl_xor(p, 8, 32);
            if ((lane & 15) == 0)                // lane 0 -> row v, lane 16 -> row v+8
                atomicAdd(&sScore[rg * 16 + (lane >> 4) * 8 + v], p);
        }
        __syncthreads();
        if (t < 32) score[slab * 32 + t] = sScore[t] + b2v;
    }
}

// order-preserving float<->uint keys so segment-max can use integer atomics
static __device__ __forceinline__ unsigned enc_key(float s) {
    unsigned u = __float_as_uint(s);
    return (u & 0x80000000u) ? ~u : (u | 0x80000000u);
}
static __device__ __forceinline__ float dec_key(unsigned k) {
    unsigned u = (k & 0x80000000u) ? (k ^ 0x80000000u) : ~k;  // k==0 -> NaN (empty)
    return __uint_as_float(u);
}

// ---------------------------------------------------------------------------
// Kernel 2: segment max of score (LDS partial max, then global)
// ---------------------------------------------------------------------------
__global__ __launch_bounds__(256) void segmax_kernel(
    const float* __restrict__ score, const int* __restrict__ group, unsigned* keys)
{
    __shared__ unsigned sk[NNET];
    const int t = threadIdx.x;
    if (t < NNET) sk[t] = 0u;
    __syncthreads();
    const int i = blockIdx.x * 256 + t;
    atomicMax(&sk[group[i]], enc_key(score[i]));
    __syncthreads();
    if (t < NNET && sk[t] != 0u) atomicMax(&keys[t], sk[t]);
}

// ---------------------------------------------------------------------------
// Kernel 3: w = exp(score - segmax), denom = segment_sum(w)
// ---------------------------------------------------------------------------
__global__ __launch_bounds__(256) void wexp_kernel(
    const float* __restrict__ score, const int* __restrict__ group,
    const unsigned* __restrict__ keys, float* __restrict__ w, float* denom)
{
    __shared__ float sd[NNET];
    const int t = threadIdx.x;
    if (t < NNET) sd[t] = 0.0f;
    __syncthreads();
    const int i = blockIdx.x * 256 + t;
    const int g = group[i];
    float m = dec_key(keys[g]);
    if (!(fabsf(m) < __builtin_inff())) m = 0.0f;   // non-finite -> 0 (ref semantics)
    const float wv = __expf(score[i] - m);
    w[i] = wv;
    atomicAdd(&sd[g], wv);
    __syncthreads();
    if (t < NNET && sd[t] != 0.0f) atomicAdd(&denom[t], sd[t]);
}

// ---------------------------------------------------------------------------
// Kernel 4: num[g][d] += w[i] * x[i][d]  (LDS 128x32 partials per block)
//   grid = 128 row-chunks x 8 col-blocks
// ---------------------------------------------------------------------------
__global__ __launch_bounds__(256) void scatter_kernel(
    const float* __restrict__ x, const int* __restrict__ group,
    const float* __restrict__ w, float* __restrict__ num)
{
    __shared__ float sacc[NNET * 33];              // padded: 33 floats per group
    const int t  = threadIdx.x;
    const int cb = blockIdx.x & 7;
    const int rc = blockIdx.x >> 3;
    const int c0 = cb * 32;
    for (int e = t; e < NNET * 33; e += 256) sacc[e] = 0.0f;
    __syncthreads();

    const int c   = t & 31;
    const int rr0 = t >> 5;
    const int base = rc * ROWS_PER;
    for (int r = rr0; r < ROWS_PER; r += 8) {
        const int row  = base + r;
        const int g    = group[row];
        const float wv = w[row];
        const float xv = x[(size_t)row * DIM + c0 + c];
        atomicAdd(&sacc[g * 33 + c], wv * xv);     // ds_add_f32
    }
    __syncthreads();
    for (int e = t; e < NNET * 32; e += 256) {
        const int g = e >> 5, cc = e & 31;
        const float v = sacc[g * 33 + cc];
        if (v != 0.0f) atomicAdd(&num[g * DIM + c0 + cc], v);
    }
}

// ---------------------------------------------------------------------------
// Kernel 5: out = denom>0 ? num/max(denom,1e-30) : 0
// ---------------------------------------------------------------------------
__global__ __launch_bounds__(256) void finalize_kernel(
    const float* __restrict__ num, const float* __restrict__ denom, float* __restrict__ out)
{
    const int i = blockIdx.x * 256 + threadIdx.x;
    const float d = denom[i >> 8];
    out[i] = (d > 0.0f) ? num[i] / fmaxf(d, 1e-30f) : 0.0f;
}

// ---------------------------------------------------------------------------
extern "C" void kernel_launch(void* const* d_in, const int* in_sizes, int n_in,
                              void* d_out, int out_size, void* d_ws, size_t ws_size,
                              hipStream_t stream)
{
    (void)in_sizes; (void)n_in; (void)out_size; (void)ws_size;

    const float* x     = (const float*)d_in[0];
    const int*   group = (const int*)  d_in[1];
    const float* W1    = (const float*)d_in[2];
    const float* b1    = (const float*)d_in[3];
    const float* W2    = (const float*)d_in[4];
    const float* b2    = (const float*)d_in[5];
    float* out = (float*)d_out;

    float*    score = (float*)d_ws;            // N_ROI
    float*    w     = score + N_ROI;           // N_ROI
    unsigned* keys  = (unsigned*)(w + N_ROI);  // NNET
    float*    denom = (float*)(keys + NNET);   // NNET
    float*    num   = denom + NNET;            // NNET*DIM

    init_kernel     <<<(NNET * DIM + 255) / 256, 256, 0, stream>>>(keys, denom, num);
    mlp_score_kernel<<<1024, 256, 0, stream>>>(x, W1, b1, W2, b2, score);
    segmax_kernel   <<<N_ROI / 256, 256, 0, stream>>>(score, group, keys);
    wexp_kernel     <<<N_ROI / 256, 256, 0, stream>>>(score, group, keys, w, denom);
    scatter_kernel  <<<(N_ROI / ROWS_PER) * 8, 256, 0, stream>>>(x, group, w, num);
    finalize_kernel <<<NNET * DIM / 256, 256, 0, stream>>>(num, denom, out);
}
